// GraphConvModel_24232205484153
// MI455X (gfx1250) — matrix operand order repaired and checked by reading, not compiled, not run on hardware
//
#include <hip/hip_runtime.h>
#include <hip/hip_bf16.h>

// Problem constants (from reference)
#define N_NODES 50000
#define N_EDGES 800000
#define N_GRAPH 64
#define F_INF   16
#define HID     512
#define OUT_DIM 24
#define N_LAYER 6

typedef __attribute__((ext_vector_type(16))) __bf16 v16bf;
typedef __attribute__((ext_vector_type(8)))  float  v8f;
typedef __attribute__((ext_vector_type(4)))  unsigned int u32x4;

union FragBF { v16bf v; u32x4 u[2]; };
union Pack8  { u32x4 u; __bf16 h[8]; };

// ------------------------------------------------------------------
// small utility kernels
// ------------------------------------------------------------------
__global__ void gc_deg(const int* __restrict__ edge, float* __restrict__ deg) {
  int e = blockIdx.x * blockDim.x + threadIdx.x;
  if (e < N_EDGES) atomicAdd(&deg[edge[N_EDGES + e]], 1.0f);
}

__global__ void gc_recip(float* __restrict__ deg) {
  int n = blockIdx.x * blockDim.x + threadIdx.x;
  if (n < N_NODES) deg[n] = 1.0f / fmaxf(deg[n], 1.0f);
}

// layer-1 scatter-add of 16-dim raw features (aggr='add')
__global__ void gc_scatter16(const float* __restrict__ x,
                             const int* __restrict__ edge,
                             float* __restrict__ agg16) {
  int idx = blockIdx.x * blockDim.x + threadIdx.x;   // E * 4
  if (idx >= N_EDGES * 4) return;
  int e = idx >> 2, c4 = idx & 3;
  int s = edge[e], d = edge[N_EDGES + e];
  float4 v = ((const float4*)x)[s * 4 + c4];
  float* ap = &agg16[d * F_INF + c4 * 4];
  atomicAdd(ap + 0, v.x); atomicAdd(ap + 1, v.y);
  atomicAdd(ap + 2, v.z); atomicAdd(ap + 3, v.w);
}

// hidden-layer scatter-add of 512-dim features (mean applied later via rdeg)
__global__ void gc_scatterH(const float* __restrict__ h,
                            const int* __restrict__ edge,
                            float* __restrict__ agg) {
  int idx = blockIdx.x * blockDim.x + threadIdx.x;   // E * 128
  int e = idx >> 7, c4 = idx & 127;
  if (e >= N_EDGES) return;
  int s = edge[e], d = edge[N_EDGES + e];
  float4 v = ((const float4*)h)[s * 128 + c4];
  float* ap = &agg[d * HID + c4 * 4];
  atomicAdd(ap + 0, v.x); atomicAdd(ap + 1, v.y);
  atomicAdd(ap + 2, v.z); atomicAdd(ap + 3, v.w);
}

// pack [Wr ; Wrt] (each Khalf x Nout fp32, row = k) into bf16, N-major: Wt[n][k]
__global__ void gc_pack_w(const float* __restrict__ wr, const float* __restrict__ wrt,
                          __bf16* __restrict__ Wt, int Khalf, int Nout) {
  int idx = blockIdx.x * blockDim.x + threadIdx.x;
  int Kc = 2 * Khalf;
  if (idx >= Nout * Kc) return;
  int n = idx / Kc, k = idx - n * Kc;
  float v = (k < Khalf) ? wr[k * Nout + n] : wrt[(k - Khalf) * Nout + n];
  Wt[(size_t)n * Kc + k] = (__bf16)v;
}

// ------------------------------------------------------------------
// fused WMMA GEMM:  Out = relu( [scale*A1 | A2] @ Wt^T + bias )
//   A1: [M][K1] fp32, A2: [M][K2] fp32, Wt: [Nout][K1+K2] bf16 (N-major)
//   block = 256 threads (8 waves), tile 128(M) x 128(N), K step 32
//   software-pipelined: next K-tile fetched to registers under the WMMAs
// ------------------------------------------------------------------
#define TM 128
#define TN 128
#define TK 32
#define APITCH 40   // bf16 elements per LDS row (32 + 8 pad)

struct ATile { Pack8 p0, p1; };
struct BTile { u32x4 b0, b1; };

// One 16-wide K segment never straddles the A1/A2 boundary (K1 % 16 == 0),
// so pick source + scale once and do 4 aligned float4 (b128) loads.
__device__ __forceinline__ void gc_fetchA(const float* __restrict__ A1, int K1,
                                          const float* __restrict__ A2, int K2,
                                          const float* __restrict__ scaleA1,
                                          int m, int M, int kg0, ATile& t) {
  if (m < M) {
    const float* src;
    float sc = 1.0f;
    if (kg0 < K1) {
      src = A1 + (size_t)m * K1 + kg0;
      if (scaleA1) sc = scaleA1[m];
    } else {
      src = A2 + (size_t)m * K2 + (kg0 - K1);
    }
    float4 f0 = ((const float4*)src)[0];
    float4 f1 = ((const float4*)src)[1];
    float4 f2 = ((const float4*)src)[2];
    float4 f3 = ((const float4*)src)[3];
    t.p0.h[0] = (__bf16)(f0.x * sc); t.p0.h[1] = (__bf16)(f0.y * sc);
    t.p0.h[2] = (__bf16)(f0.z * sc); t.p0.h[3] = (__bf16)(f0.w * sc);
    t.p0.h[4] = (__bf16)(f1.x * sc); t.p0.h[5] = (__bf16)(f1.y * sc);
    t.p0.h[6] = (__bf16)(f1.z * sc); t.p0.h[7] = (__bf16)(f1.w * sc);
    t.p1.h[0] = (__bf16)(f2.x * sc); t.p1.h[1] = (__bf16)(f2.y * sc);
    t.p1.h[2] = (__bf16)(f2.z * sc); t.p1.h[3] = (__bf16)(f2.w * sc);
    t.p1.h[4] = (__bf16)(f3.x * sc); t.p1.h[5] = (__bf16)(f3.y * sc);
    t.p1.h[6] = (__bf16)(f3.z * sc); t.p1.h[7] = (__bf16)(f3.w * sc);
  } else {
    #pragma unroll
    for (int i = 0; i < 8; ++i) { t.p0.h[i] = (__bf16)0.0f; t.p1.h[i] = (__bf16)0.0f; }
  }
}

__device__ __forceinline__ void gc_fetchB(const __bf16* __restrict__ Wt,
                                          size_t rowbase, int k0, BTile& t) {
  const u32x4* s = (const u32x4*)(Wt + rowbase + k0);
  t.b0 = s[0];
  t.b1 = s[1];
}

__global__ __launch_bounds__(256)
void gc_gemm_wmma(const float* __restrict__ A1, int K1,
                  const float* __restrict__ A2, int K2,
                  const float* __restrict__ scaleA1,   // per-row scale for A1 (nullptr = 1.0)
                  const __bf16* __restrict__ Wt,       // [Nout][Kc] bf16
                  const float* __restrict__ bias,      // [Nout]
                  float* __restrict__ Out, int M, int Nout) {
  __shared__ __bf16 Als[TM * APITCH];
  __shared__ __bf16 Bls[TN * APITCH];

  const int Kc = K1 + K2;
  const int m0 = blockIdx.x * TM;
  const int n0 = blockIdx.y * TN;
  const int tid  = threadIdx.x;
  const int lane = tid & 31;
  const int wid  = tid >> 5;
  const int wm = wid >> 2;          // 0..1  (wave row)
  const int wn = wid & 3;           // 0..3  (wave col)
  const int l16   = lane & 15;
  const int lhalf = lane >> 4;

  v8f acc[4][2];
  #pragma unroll
  for (int t = 0; t < 4; ++t)
    #pragma unroll
    for (int u = 0; u < 2; ++u)
      #pragma unroll
      for (int i = 0; i < 8; ++i) acc[t][u][i] = 0.0f;

  const int srow = tid >> 1;        // 0..127 : staged row (A: m, B: n)
  const int sseg = tid & 1;         // 0..1   : 16-element half of the 32-K row
  const int m    = m0 + srow;
  const size_t browbase = (size_t)(n0 + srow) * Kc + sseg * 16;
  const int ldsoff = srow * APITCH + sseg * 16;

  // prologue: fetch first K-tile into registers
  ATile at; BTile bt;
  gc_fetchA(A1, K1, A2, K2, scaleA1, m, M, sseg * 16, at);
  gc_fetchB(Wt, browbase, 0, bt);

  for (int k0 = 0; k0 < Kc; k0 += TK) {
    // ---- commit staged registers to LDS ----
    *(u32x4*)&Als[ldsoff]     = at.p0.u;
    *(u32x4*)&Als[ldsoff + 8] = at.p1.u;
    *(u32x4*)&Bls[ldsoff]     = bt.b0;
    *(u32x4*)&Bls[ldsoff + 8] = bt.b1;
    __syncthreads();

    // ---- issue next tile's global loads (hide under WMMAs) ----
    const int kn = k0 + TK;
    if (kn < Kc) {
      gc_fetchA(A1, K1, A2, K2, scaleA1, m, M, kn + sseg * 16, at);
      gc_fetchB(Wt, browbase, kn, bt);
    }

    // ---- fragments + 8 x v_wmma_f32_16x16x32_bf16 ----
    FragBF fa[4], fb[2];
    #pragma unroll
    for (int t = 0; t < 4; ++t) {
      int mr = wm * 64 + t * 16 + l16;
      fa[t].u[0] = *(const u32x4*)&Als[mr * APITCH + lhalf * 8];
      fa[t].u[1] = *(const u32x4*)&Als[mr * APITCH + 16 + lhalf * 8];
    }
    #pragma unroll
    for (int u = 0; u < 2; ++u) {
      int nr = wn * 32 + u * 16 + l16;
      fb[u].u[0] = *(const u32x4*)&Bls[nr * APITCH + lhalf * 8];
      fb[u].u[1] = *(const u32x4*)&Bls[nr * APITCH + 16 + lhalf * 8];
    }
    #pragma unroll
    for (int t = 0; t < 4; ++t)
      #pragma unroll
      for (int u = 0; u < 2; ++u)
        acc[t][u] = __builtin_amdgcn_wmma_f32_16x16x32_bf16(
            false, fa[t].v, false, fb[u].v, (short)0, acc[t][u], false, false);

    __syncthreads();
  }

  // ---- epilogue: +bias, relu, store ----
  #pragma unroll
  for (int t = 0; t < 4; ++t) {
    #pragma unroll
    for (int u = 0; u < 2; ++u) {
      int col = n0 + wn * 32 + u * 16 + l16;
      float b = bias[col];
      #pragma unroll
      for (int r = 0; r < 8; ++r) {
        int row = m0 + wm * 64 + t * 16 + lhalf * 8 + r;
        if (row < M) {
          float v = acc[t][u][r] + b;
          Out[(size_t)row * Nout + col] = v > 0.0f ? v : 0.0f;
        }
      }
    }
  }
}

// ------------------------------------------------------------------
// pooling + output head
// ------------------------------------------------------------------
__global__ void gc_count(const int* __restrict__ batch, float* __restrict__ counts) {
  int n = blockIdx.x * blockDim.x + threadIdx.x;
  if (n < N_NODES) atomicAdd(&counts[batch[n]], 1.0f);
}

__global__ void gc_pool(const float* __restrict__ h, const int* __restrict__ batch,
                        float* __restrict__ pooled) {
  int idx = blockIdx.x * blockDim.x + threadIdx.x;   // N * 128
  int n = idx >> 7, c4 = idx & 127;
  if (n >= N_NODES) return;
  int g = batch[n];
  float4 v = ((const float4*)h)[n * 128 + c4];
  float* pp = &pooled[g * HID + c4 * 4];
  atomicAdd(pp + 0, v.x); atomicAdd(pp + 1, v.y);
  atomicAdd(pp + 2, v.z); atomicAdd(pp + 3, v.w);
}

__global__ void gc_pooldiv(float* __restrict__ pooled, const float* __restrict__ counts) {
  int i = blockIdx.x * blockDim.x + threadIdx.x;
  if (i < N_GRAPH * HID) pooled[i] *= 1.0f / fmaxf(counts[i >> 9], 1.0f);
}

__global__ void gc_out(const float* __restrict__ pooled, const float* __restrict__ w_out,
                       const float* __restrict__ b_out, float* __restrict__ out) {
  int t = blockIdx.x * blockDim.x + threadIdx.x;
  if (t >= N_GRAPH * OUT_DIM) return;
  int g = t / OUT_DIM, o = t - g * OUT_DIM;
  float acc = b_out[o];
  for (int k = 0; k < HID; ++k)
    acc += pooled[g * HID + k] * w_out[k * OUT_DIM + o];
  out[t] = acc;
}

// ------------------------------------------------------------------
// host orchestration
// ------------------------------------------------------------------
extern "C" void kernel_launch(void* const* d_in, const int* in_sizes, int n_in,
                              void* d_out, int out_size, void* d_ws, size_t ws_size,
                              hipStream_t stream) {
  const float* x       = (const float*)d_in[0];
  const int*   edge    = (const int*)  d_in[1];   // [2][E]
  const int*   batch   = (const int*)  d_in[2];   // [N]
  const float* w_rel1  = (const float*)d_in[3];   // [16][512]
  const float* b_rel1  = (const float*)d_in[4];   // [512]
  const float* w_root1 = (const float*)d_in[5];   // [16][512]
  const float* w_rel   = (const float*)d_in[6];   // [6][512][512]
  const float* b_rel   = (const float*)d_in[7];   // [6][512]
  const float* w_root  = (const float*)d_in[8];   // [6][512][512]
  const float* w_out   = (const float*)d_in[9];   // [512][24]
  const float* b_out   = (const float*)d_in[10];  // [24]
  float* out = (float*)d_out;

  // workspace carve-out (256B aligned slices)
  char* p = (char*)d_ws;
  auto alloc = [&](size_t bytes) -> void* {
    void* r = (void*)p; p += (bytes + 255) & ~(size_t)255; return r;
  };
  const size_t hbytes = (size_t)N_NODES * HID * sizeof(float);
  float*  hA     = (float*) alloc(hbytes);
  float*  hB     = (float*) alloc(hbytes);
  float*  agg    = (float*) alloc(hbytes);
  float*  agg16  = (float*) alloc((size_t)N_NODES * F_INF * sizeof(float));
  float*  rdeg   = (float*) alloc((size_t)N_NODES * sizeof(float));
  __bf16* Wt     = (__bf16*)alloc((size_t)HID * 2 * HID * sizeof(__bf16));
  float*  pooled = (float*) alloc((size_t)N_GRAPH * HID * sizeof(float));
  float*  counts = (float*) alloc((size_t)N_GRAPH * sizeof(float));
  if ((size_t)(p - (char*)d_ws) > ws_size) return;  // scratch too small: bail

  const dim3 gemm_grid((N_NODES + TM - 1) / TM, HID / TN);

  // degrees -> reciprocal (for mean aggregation)
  hipMemsetAsync(rdeg, 0, (size_t)N_NODES * sizeof(float), stream);
  gc_deg  <<<(N_EDGES + 255) / 256, 256, 0, stream>>>(edge, rdeg);
  gc_recip<<<(N_NODES + 255) / 256, 256, 0, stream>>>(rdeg);

  // ---- layer 1: relu( (sum-agg x) @ Wrel1 + x @ Wroot1 + b ) ----
  hipMemsetAsync(agg16, 0, (size_t)N_NODES * F_INF * sizeof(float), stream);
  gc_scatter16<<<(N_EDGES * 4 + 255) / 256, 256, 0, stream>>>(x, edge, agg16);
  gc_pack_w<<<(HID * 2 * F_INF + 255) / 256, 256, 0, stream>>>(w_rel1, w_root1, Wt, F_INF, HID);
  gc_gemm_wmma<<<gemm_grid, 256, 0, stream>>>(agg16, F_INF, x, F_INF, nullptr,
                                              Wt, b_rel1, hA, N_NODES, HID);

  // ---- layers 2..7: relu( (mean-agg h) @ Wr + h @ Wrt + b ) ----
  float* hin = hA; float* hout = hB;
  for (int l = 0; l < N_LAYER; ++l) {
    hipMemsetAsync(agg, 0, hbytes, stream);
    gc_scatterH<<<(N_EDGES * 128) / 256, 256, 0, stream>>>(hin, edge, agg);
    gc_pack_w<<<(HID * 2 * HID + 255) / 256, 256, 0, stream>>>(
        w_rel + (size_t)l * HID * HID, w_root + (size_t)l * HID * HID, Wt, HID, HID);
    gc_gemm_wmma<<<gemm_grid, 256, 0, stream>>>(agg, HID, hin, HID, rdeg,
                                                Wt, b_rel + (size_t)l * HID, hout,
                                                N_NODES, HID);
    float* tmp = hin; hin = hout; hout = tmp;
  }

  // ---- global mean pool + output linear ----
  hipMemsetAsync(pooled, 0, (size_t)N_GRAPH * HID * sizeof(float), stream);
  hipMemsetAsync(counts, 0, (size_t)N_GRAPH * sizeof(float), stream);
  gc_count<<<(N_NODES + 255) / 256, 256, 0, stream>>>(batch, counts);
  gc_pool <<<(N_NODES * 128 + 255) / 256, 256, 0, stream>>>(hin, batch, pooled);
  gc_pooldiv<<<(N_GRAPH * HID + 255) / 256, 256, 0, stream>>>(pooled, counts);
  gc_out<<<(N_GRAPH * OUT_DIM + 255) / 256, 256, 0, stream>>>(pooled, w_out, b_out, out);
}
